// BIDAF_14568529068138
// MI455X (gfx1250) — compile-verified, matrix operand order checked by reading, not run on hardware
//
#include <hip/hip_runtime.h>
#include <hip/hip_bf16.h>

#define B_      32
#define P_      4096
#define Q_      64
#define D_      128
#define PTILE   64
#define NTILE   (P_ / PTILE)     // 64 tiles per batch
#define NEG_INF (-1e30f)

typedef __attribute__((ext_vector_type(16))) _Float16 v16h;
typedef __attribute__((ext_vector_type(8)))  float    v8f;

// ---------------- WMMA fragment gathers (wave32, 16x16x32 f16) ----------------
// A (16x32, MxK) / B (32x16, KxN) lane layout: lane&15 selects row (A) / col (B),
// hi = lane>>4: halves 0..7 -> K = k0 + 8*hi + j ; halves 8..15 -> K = k0 + 16 + 8*hi + j.

__device__ __forceinline__ v16h frag_rm(const _Float16* base, int row, int ld,
                                        int k0, int hi) {
  v16h f;
  const _Float16* r = base + row * ld + k0 + 8 * hi;
#pragma unroll
  for (int j = 0; j < 8; ++j) f[j] = r[j];
#pragma unroll
  for (int j = 0; j < 8; ++j) f[8 + j] = r[16 + j];
  return f;
}

__device__ __forceinline__ v16h frag_rm_f32(const float* base, int row, int ld,
                                            int k0, int hi) {
  v16h f;
  const float* r = base + row * ld + k0 + 8 * hi;
#pragma unroll
  for (int j = 0; j < 8; ++j) f[j] = (_Float16)r[j];
#pragma unroll
  for (int j = 0; j < 8; ++j) f[8 + j] = (_Float16)r[16 + j];
  return f;
}

// column gather: element(k, n) = base[(k0+K)*ld + col]
__device__ __forceinline__ v16h frag_cm(const _Float16* base, int col, int ld,
                                        int k0, int hi) {
  v16h f;
  int kb = k0 + 8 * hi;
#pragma unroll
  for (int j = 0; j < 8; ++j) f[j] = base[(kb + j) * ld + col];
#pragma unroll
  for (int j = 0; j < 8; ++j) f[8 + j] = base[(kb + 16 + j) * ld + col];
  return f;
}

__device__ __forceinline__ void atomicMaxFloat(float* addr, float v) {
  if (v >= 0.0f)
    atomicMax((int*)addr, __float_as_int(v));
  else
    atomicMin((unsigned int*)addr, __float_as_uint(v));
}

// ---------------- kernels ----------------

__global__ void k_init(float* pooled, int n) {
  int i = blockIdx.x * blockDim.x + threadIdx.x;
  if (i < n) pooled[i] = NEG_INF;
}

// V[b,q,d] = sum_e W[d,e] * U[b,q,e]   (bilinear folded into U side: 67 MFLOP)
__global__ __launch_bounds__(128) void k_V(const float* __restrict__ U,
                                           const float* __restrict__ W,
                                           float* __restrict__ V) {
  __shared__ float u[D_];
  int b = blockIdx.x / Q_, q = blockIdx.x % Q_;
  int t = threadIdx.x;
  u[t] = U[((size_t)b * Q_ + q) * D_ + t];
  __syncthreads();
  const float* wr = W + t * D_;
  float acc = 0.0f;
#pragma unroll 8
  for (int e = 0; e < D_; ++e) acc += wr[e] * u[e];
  V[((size_t)b * Q_ + q) * D_ + t] = acc;
}

// Pass 1: S tile via WMMA, rowmax over q, streaming-softmax partials over p.
__global__ __launch_bounds__(128) void k_pass1(const float* __restrict__ H,
                                               const float* __restrict__ V,
                                               float* __restrict__ part_m,
                                               float* __restrict__ part_s,
                                               float* __restrict__ part_q2c) {
  __shared__ _Float16 Hlds[PTILE * D_];  // 16 KB
  __shared__ _Float16 Vlds[Q_ * D_];     // 16 KB
  __shared__ float    Slds[PTILE * Q_];  // 16 KB
  __shared__ float    rowred[PTILE];
  __shared__ float    tile_ms[2];

  int blk = blockIdx.x;
  int b = blk / NTILE, tile = blk % NTILE;
  int p0 = tile * PTILE;
  int t = threadIdx.x;
  const float* Hb = H + ((size_t)b * P_ + p0) * D_;
  const float* Vb = V + (size_t)b * Q_ * D_;

  for (int i = t; i < PTILE * D_; i += 128) Hlds[i] = (_Float16)Hb[i];
  for (int i = t; i < Q_ * D_;   i += 128) Vlds[i] = (_Float16)Vb[i];
  __syncthreads();

  int wv = t >> 5, lane = t & 31;
  int hi = lane >> 4, lo = lane & 15;
  int rowbase = wv * 16;

  for (int qt = 0; qt < 4; ++qt) {
    v8f acc = {};
#pragma unroll
    for (int kc = 0; kc < 4; ++kc) {
      v16h a  = frag_rm(Hlds, rowbase + lo, D_, kc * 32, hi);
      v16h bm = frag_rm(Vlds, qt * 16 + lo, D_, kc * 32, hi);  // B = V^T
      acc = __builtin_amdgcn_wmma_f32_16x16x32_f16(false, a, false, bm,
                                                   (short)0, acc, false, false);
    }
#pragma unroll
    for (int r = 0; r < 8; ++r)
      Slds[(rowbase + r + 8 * hi) * Q_ + qt * 16 + lo] = acc[r];
  }
  __syncthreads();

  if (t < PTILE) {
    const float* sr = Slds + t * Q_;
    float mx = sr[0];
    for (int q = 1; q < Q_; ++q) mx = fmaxf(mx, sr[q]);
    rowred[t] = mx;
  }
  __syncthreads();
  if (t == 0) {
    float mt = rowred[0];
    for (int r = 1; r < PTILE; ++r) mt = fmaxf(mt, rowred[r]);
    float s = 0.0f;
    for (int r = 0; r < PTILE; ++r) {
      float w = __expf(rowred[r] - mt);
      rowred[r] = w;  // overwrite with weight
      s += w;
    }
    tile_ms[0] = mt;
    tile_ms[1] = s;
  }
  __syncthreads();

  // partial q2c: sum_p w_p * H[p, d]   (H rows are WGP$/L2 hot)
  {
    float acc = 0.0f;
    for (int r = 0; r < PTILE; ++r) acc += rowred[r] * Hb[r * D_ + t];
    part_q2c[((size_t)b * NTILE + tile) * D_ + t] = acc;
  }
  if (t == 0) {
    part_m[b * NTILE + tile] = tile_ms[0];
    part_s[b * NTILE + tile] = tile_ms[1];
  }
}

// Merge streaming-softmax partials -> q2c[b, d]
__global__ __launch_bounds__(128) void k_merge(const float* __restrict__ part_m,
                                               const float* __restrict__ part_s,
                                               const float* __restrict__ part_q2c,
                                               float* __restrict__ q2c) {
  __shared__ float Ms[2];
  int b = blockIdx.x, t = threadIdx.x;
  if (t == 0) {
    float M = part_m[b * NTILE];
    for (int i = 1; i < NTILE; ++i) M = fmaxf(M, part_m[b * NTILE + i]);
    float s = 0.0f;
    for (int i = 0; i < NTILE; ++i)
      s += part_s[b * NTILE + i] * __expf(part_m[b * NTILE + i] - M);
    Ms[0] = M;
    Ms[1] = s;
  }
  __syncthreads();
  float M = Ms[0], s = Ms[1];
  float acc = 0.0f;
  for (int i = 0; i < NTILE; ++i)
    acc += part_q2c[((size_t)b * NTILE + i) * D_ + t] *
           __expf(part_m[b * NTILE + i] - M);
  q2c[b * D_ + t] = acc / s;
}

// Pass 2: recompute S (WMMA), softmax_q, c2q (WMMA), masked max-pool.
__global__ __launch_bounds__(128) void k_pass2(const float* __restrict__ H,
                                               const float* __restrict__ U,
                                               const float* __restrict__ Mmat,
                                               const int*   __restrict__ words,
                                               const float* __restrict__ V,
                                               const float* __restrict__ q2c,
                                               float* __restrict__ pooled) {
  __shared__ _Float16 Hlds[PTILE * D_];   // 16 KB
  __shared__ _Float16 VUlds[Q_ * D_];     // 16 KB : V first, restaged with U
  __shared__ float    Slds[PTILE * Q_];   // 16 KB : S -> probs, then aliased as c2q f16
  _Float16* c2qlds = (_Float16*)Slds;     // 64 x 128 f16 = 16 KB (reuses Slds)

  int blk = blockIdx.x;
  int b = blk / NTILE, tile = blk % NTILE;
  int p0 = tile * PTILE;
  int t = threadIdx.x;
  const float* Hb = H    + ((size_t)b * P_ + p0) * D_;
  const float* Mb = Mmat + ((size_t)b * P_ + p0) * D_;
  const float* Ub = U    + (size_t)b * Q_ * D_;
  const float* Vb = V    + (size_t)b * Q_ * D_;

  for (int i = t; i < PTILE * D_; i += 128) Hlds[i]  = (_Float16)Hb[i];
  for (int i = t; i < Q_ * D_;   i += 128) VUlds[i] = (_Float16)Vb[i];
  __builtin_prefetch(Mb + t, 0, 3);  // warm M column into cache (global_prefetch_b8)
  __syncthreads();

  int wv = t >> 5, lane = t & 31;
  int hi = lane >> 4, lo = lane & 15;
  int rowbase = wv * 16;

  // ---- S = H (f16) x V^T (f16), f32 accum ----
  for (int qt = 0; qt < 4; ++qt) {
    v8f acc = {};
#pragma unroll
    for (int kc = 0; kc < 4; ++kc) {
      v16h a  = frag_rm(Hlds,  rowbase + lo, D_, kc * 32, hi);
      v16h bm = frag_rm(VUlds, qt * 16 + lo, D_, kc * 32, hi);
      acc = __builtin_amdgcn_wmma_f32_16x16x32_f16(false, a, false, bm,
                                                   (short)0, acc, false, false);
    }
#pragma unroll
    for (int r = 0; r < 8; ++r)
      Slds[(rowbase + r + 8 * hi) * Q_ + qt * 16 + lo] = acc[r];
  }
  __syncthreads();

  // ---- softmax over q (rows), in place; restage U into VUlds ----
  if (t < PTILE) {
    float* sr = Slds + t * Q_;
    float mx = sr[0];
    for (int q = 1; q < Q_; ++q) mx = fmaxf(mx, sr[q]);
    float s = 0.0f;
    for (int q = 0; q < Q_; ++q) {
      float e = __expf(sr[q] - mx);
      sr[q] = e;
      s += e;
    }
    float inv = 1.0f / s;
    for (int q = 0; q < Q_; ++q) sr[q] *= inv;
  }
  for (int i = t; i < Q_ * D_; i += 128) VUlds[i] = (_Float16)Ub[i];
  __syncthreads();

  // ---- c2q = probs (f16) x U (f16) ; keep all 8 d-tiles in registers ----
  v8f cacc[8];
#pragma unroll
  for (int dt = 0; dt < 8; ++dt) {
    v8f acc = {};
#pragma unroll
    for (int kc = 0; kc < 2; ++kc) {
      v16h a  = frag_rm_f32(Slds, rowbase + lo, Q_, kc * 32, hi);
      v16h bm = frag_cm(VUlds, dt * 16 + lo, D_, kc * 32, hi);
      acc = __builtin_amdgcn_wmma_f32_16x16x32_f16(false, a, false, bm,
                                                   (short)0, acc, false, false);
    }
    cacc[dt] = acc;
  }
  __syncthreads();  // all probs reads done -> safe to alias Slds as c2q f16
#pragma unroll
  for (int dt = 0; dt < 8; ++dt)
#pragma unroll
    for (int r = 0; r < 8; ++r)
      c2qlds[(rowbase + r + 8 * hi) * D_ + dt * 16 + lo] = (_Float16)cacc[dt][r];
  __syncthreads();

  // ---- masked max-pool over the 64 rows of this tile; thread t = column d ----
  float qcv = q2c[b * D_ + t];
  float m0 = NEG_INF, m1 = NEG_INF, m2 = NEG_INF, m3 = NEG_INF, m4 = NEG_INF;
  const int* wrow = words + (size_t)b * P_ + p0;
  for (int r = 0; r < PTILE; ++r) {
    int rp = (r + 4 < PTILE) ? (r + 4) : (PTILE - 1);
    __builtin_prefetch(&Mb[rp * D_ + t], 0, 0);
    if (wrow[r] == 0) continue;  // PAD -> contributes NEG_INF
    float h  = Hb[r * D_ + t];
    float c  = (float)c2qlds[r * D_ + t];
    float mv = Mb[r * D_ + t];
    m0 = fmaxf(m0, h);
    m1 = fmaxf(m1, c);
    m2 = fmaxf(m2, h * c);
    m3 = fmaxf(m3, h * qcv);
    m4 = fmaxf(m4, mv);
  }
  float* pb = pooled + (size_t)b * 5 * D_;
  atomicMaxFloat(pb + 0 * D_ + t, m0);
  atomicMaxFloat(pb + 1 * D_ + t, m1);
  atomicMaxFloat(pb + 2 * D_ + t, m2);
  atomicMaxFloat(pb + 3 * D_ + t, m3);
  atomicMaxFloat(pb + 4 * D_ + t, m4);
}

// out[b, j] = pooled[b, :] . W_cls[:, j]
__global__ __launch_bounds__(64) void k_cls(const float* __restrict__ pooled,
                                            const float* __restrict__ Wc,
                                            float* __restrict__ out) {
  __shared__ float red[64];
  int b = blockIdx.x, t = threadIdx.x;
  float a0 = 0.0f, a1 = 0.0f;
  for (int k = t; k < 5 * D_; k += 64) {
    float p = pooled[(size_t)b * 5 * D_ + k];
    a0 += p * Wc[k * 2 + 0];
    a1 += p * Wc[k * 2 + 1];
  }
  red[t] = a0;
  __syncthreads();
  if (t == 0) {
    float s = 0.0f;
    for (int i = 0; i < 64; ++i) s += red[i];
    out[b * 2 + 0] = s;
  }
  __syncthreads();
  red[t] = a1;
  __syncthreads();
  if (t == 0) {
    float s = 0.0f;
    for (int i = 0; i < 64; ++i) s += red[i];
    out[b * 2 + 1] = s;
  }
}

// ---------------- launcher ----------------
extern "C" void kernel_launch(void* const* d_in, const int* in_sizes, int n_in,
                              void* d_out, int out_size, void* d_ws, size_t ws_size,
                              hipStream_t stream) {
  const float* H     = (const float*)d_in[0];  // (B, P, D)
  const float* U     = (const float*)d_in[1];  // (B, Q, D)
  const float* Mmat  = (const float*)d_in[2];  // (B, P, D)
  const int*   words = (const int*)  d_in[3];  // (B, P)
  const float* Wattn = (const float*)d_in[4];  // (D, D)
  const float* Wcls  = (const float*)d_in[5];  // (5D, 2)
  float* out = (float*)d_out;                  // (B, 2)

  // workspace carve (floats): ~2.2 MB total
  float* ws       = (float*)d_ws;
  float* V        = ws;                                   // B*Q*D
  float* part_m   = V + (size_t)B_ * Q_ * D_;             // B*NTILE
  float* part_s   = part_m + (size_t)B_ * NTILE;          // B*NTILE
  float* part_q2c = part_s + (size_t)B_ * NTILE;          // B*NTILE*D
  float* q2c      = part_q2c + (size_t)B_ * NTILE * D_;   // B*D
  float* pooled   = q2c + (size_t)B_ * D_;                // B*5*D

  int npool = B_ * 5 * D_;
  k_init <<<(npool + 255) / 256, 256, 0, stream>>>(pooled, npool);
  k_V    <<<B_ * Q_,      128, 0, stream>>>(U, Wattn, V);
  k_pass1<<<B_ * NTILE,   128, 0, stream>>>(H, V, part_m, part_s, part_q2c);
  k_merge<<<B_,           128, 0, stream>>>(part_m, part_s, part_q2c, q2c);
  k_pass2<<<B_ * NTILE,   128, 0, stream>>>(H, U, Mmat, words, V, q2c, pooled);
  k_cls  <<<B_,            64, 0, stream>>>(pooled, Wcls, out);
}
